// VectorQuantizer2_11166914969840
// MI455X (gfx1250) — matrix operand order repaired and checked by reading, not compiled, not run on hardware
//
#include <hip/hip_runtime.h>
#include <hip/hip_bf16.h>

// ---------------------------------------------------------------------------
// VQ-VAE vector quantizer for MI455X (gfx1250, wave32, WMMA).
//
// Heavy op: D[n, code] = ||e_code||^2 - 2 * z_n . e_code   (16384 x 8192, K=256)
// computed with v_wmma_f32_16x16x32_bf16 using a bf16x3 split-precision GEMM
// (zh*eh + zh*el + zl*eh in 3 independent fp32 accumulators for ILP).
// Codebook tiles are double-buffered through LDS; staging uses the CDNA5
// async DMA path (GLOBAL_LOAD_ASYNC_TO_LDS_B128 + s_wait_asynccnt) when the
// toolchain exposes it, else a synchronous copy fallback.
// ---------------------------------------------------------------------------

typedef __attribute__((ext_vector_type(16))) __bf16 v16bf;
typedef __attribute__((ext_vector_type(8)))  __bf16 v8bf;
typedef __attribute__((ext_vector_type(8)))  float  v8f;
typedef __attribute__((ext_vector_type(4)))  int    v4i;

#define N_PIX   16384      // 16 * 32 * 32
#define N_CODE  8192
#define EDIM    256
#define NB      32         // codes staged per LDS buffer
#define STAGE_BYTES (NB * EDIM * 4)   // hi (16KB) + lo (16KB) = 32KB per buffer
#define ZQ_ELEMS 4194304   // 16*256*32*32

#if defined(__has_builtin)
#if __has_builtin(__builtin_amdgcn_global_load_async_to_lds_b128)
#define VQ_HAS_ASYNC 1
#endif
#endif
#ifndef VQ_HAS_ASYNC
#define VQ_HAS_ASYNC 0
#endif

// workspace layout (bytes)
static constexpr size_t WS_ZH  = 0;                                   // bf16 [16384][256]
static constexpr size_t WS_ZL  = WS_ZH + (size_t)N_PIX * EDIM * 2;    // bf16 [16384][256]
static constexpr size_t WS_EH  = WS_ZL + (size_t)N_PIX * EDIM * 2;    // bf16 [8192][256]
static constexpr size_t WS_EL  = WS_EH + (size_t)N_CODE * EDIM * 2;   // bf16 [8192][256]
static constexpr size_t WS_NRM = WS_EL + (size_t)N_CODE * EDIM * 2;   // f32  [8192]
static constexpr size_t WS_IDX = WS_NRM + (size_t)N_CODE * 4;         // i32  [16384]
static constexpr size_t WS_CNT = WS_IDX + (size_t)N_PIX * 4;          // f32  [8192]
static constexpr size_t WS_SSE = WS_CNT + (size_t)N_CODE * 4;         // f32  [1]

// --- prep: NCHW z -> row-major zf[n][c], split into bf16 hi/lo ---------------
__global__ void vq_convert_z(const float* __restrict__ z,
                             __bf16* __restrict__ zh, __bf16* __restrict__ zl) {
    int t = blockIdx.x * blockDim.x + threadIdx.x;      // 0 .. 4194303 (NCHW flat)
    float v = z[t];
    int b = t >> 18;            // / (256*1024)
    int r = t & 262143;
    int c = r >> 10;            // channel
    int p = r & 1023;           // h*32 + w
    int n = (b << 10) + p;      // pixel row
    __bf16 hi = (__bf16)v;
    __bf16 lo = (__bf16)(v - (float)hi);
    zh[(size_t)n * EDIM + c] = hi;
    zl[(size_t)n * EDIM + c] = lo;
}

// --- prep: emb -> bf16 hi/lo -------------------------------------------------
__global__ void vq_convert_emb(const float* __restrict__ e,
                               __bf16* __restrict__ eh, __bf16* __restrict__ el) {
    int t = blockIdx.x * blockDim.x + threadIdx.x;      // 0 .. 2097151
    float v = e[t];
    __bf16 hi = (__bf16)v;
    eh[t] = hi;
    el[t] = (__bf16)(v - (float)hi);
}

// --- prep: ||e_code||^2 in fp32 ---------------------------------------------
__global__ void vq_norm_emb(const float* __restrict__ e, float* __restrict__ nrm) {
    int c = blockIdx.x * blockDim.x + threadIdx.x;
    if (c < N_CODE) {
        const float* row = e + (size_t)c * EDIM;
        float s = 0.f;
        for (int k = 0; k < EDIM; ++k) s = fmaf(row[k], row[k], s);
        nrm[c] = s;
    }
}

// --- prep: zero histogram + sse accumulator ---------------------------------
__global__ void vq_init(float* __restrict__ cnt, float* __restrict__ sse) {
    int t = blockIdx.x * blockDim.x + threadIdx.x;
    if (t < N_CODE) cnt[t] = 0.f;
    if (t == 0) sse[0] = 0.f;
}

__device__ __forceinline__ v16bf vq_load16(const __bf16* p) {
    v8bf x0 = *(const v8bf*)(p);
    v8bf x1 = *(const v8bf*)(p + 16);
    v16bf r;
#pragma unroll
    for (int i = 0; i < 8; ++i) { r[i] = x0[i]; r[8 + i] = x1[i]; }
    return r;
}

// --- main WMMA argmin kernel -------------------------------------------------
// grid: 128 blocks x 256 threads (8 waves). Each wave owns 16 z-rows kept
// resident in VGPRs; the block double-buffers 32-code B tiles (hi+lo) in LDS.
__global__ __launch_bounds__(256) void vq_argmin(
    const __bf16* __restrict__ zh, const __bf16* __restrict__ zl,
    const __bf16* __restrict__ eh, const __bf16* __restrict__ el,
    const float*  __restrict__ nrm, int* __restrict__ idx_out)
{
    extern __shared__ __align__(32) char smem[];   // 2 * STAGE_BYTES = 64KB

    const int lane = threadIdx.x & 31;
    const int wave = threadIdx.x >> 5;
    const int row_base = (blockIdx.x * 8 + wave) * 16;
    const int m    = lane & 15;
    const int half = lane >> 4;

    // ---- load resident A fragments (16 rows x 256, hi + lo) ----
    // 16-bit A 16x32 layout: lane (m, half) holds A[m, k0+8*half .. +7] then
    // A[m, k0+16+8*half .. +7]  (ISA 7.12.2)
    v16bf ah[8], al[8];
    {
        const __bf16* zrh = zh + (size_t)(row_base + m) * EDIM;
        const __bf16* zrl = zl + (size_t)(row_base + m) * EDIM;
#pragma unroll
        for (int kc = 0; kc < 8; ++kc) {
            const int k0 = kc * 32;
            v8bf h0 = *(const v8bf*)(zrh + k0 + 8 * half);
            v8bf h1 = *(const v8bf*)(zrh + k0 + 16 + 8 * half);
            v8bf l0 = *(const v8bf*)(zrl + k0 + 8 * half);
            v8bf l1 = *(const v8bf*)(zrl + k0 + 16 + 8 * half);
#pragma unroll
            for (int i = 0; i < 8; ++i) {
                ah[kc][i] = h0[i]; ah[kc][8 + i] = h1[i];
                al[kc][i] = l0[i]; al[kc][8 + i] = l1[i];
            }
        }
    }

    float minv[8];
    int   mini[8];
#pragma unroll
    for (int r = 0; r < 8; ++r) { minv[r] = 3.4e38f; mini[r] = 0; }

    // ---- B tile staging: async DMA to LDS if available, else sync copy ----
    auto issue_stage = [&](int nb, int buf) {
        char* dst = smem + buf * STAGE_BYTES;           // [hi 16KB | lo 16KB]
        const char* gh = (const char*)(eh + (size_t)nb * EDIM);
        const char* gl = (const char*)(el + (size_t)nb * EDIM);
#if VQ_HAS_ASYNC
#pragma unroll
        for (int k = 0; k < 4; ++k) {                   // 8 async b128 per wave
            const int c16 = (threadIdx.x + k * 256) * 16;
            __builtin_amdgcn_global_load_async_to_lds_b128(
                (v4i*)(uintptr_t)(gh + c16), (v4i*)(uintptr_t)(dst + c16), 0, 0);
            __builtin_amdgcn_global_load_async_to_lds_b128(
                (v4i*)(uintptr_t)(gl + c16), (v4i*)(uintptr_t)(dst + 16384 + c16), 0, 0);
        }
#else
        const uint4* gh4 = (const uint4*)gh;
        const uint4* gl4 = (const uint4*)gl;
        uint4* dh4 = (uint4*)dst;
        uint4* dl4 = (uint4*)(dst + 16384);
#pragma unroll
        for (int k = 0; k < 4; ++k) {
            const int i = threadIdx.x + k * 256;
            dh4[i] = gh4[i];
            dl4[i] = gl4[i];
        }
        if (nb + NB < N_CODE) {   // prefetch next stage (global_prefetch_b8)
            __builtin_prefetch((const char*)(eh + (size_t)(nb + NB) * EDIM) + threadIdx.x * 64, 0, 1);
            __builtin_prefetch((const char*)(el + (size_t)(nb + NB) * EDIM) + threadIdx.x * 64, 0, 1);
        }
#endif
    };

    issue_stage(0, 0);

    for (int s = 0; s < N_CODE / NB; ++s) {
        const int nb = s * NB;
        if (nb + NB < N_CODE) {
            issue_stage(nb + NB, (s + 1) & 1);
#if VQ_HAS_ASYNC
            // async loads complete in order: <=8 outstanding -> stage s done
            asm volatile("s_wait_asynccnt 0x8" ::: "memory");
#endif
        } else {
#if VQ_HAS_ASYNC
            asm volatile("s_wait_asynccnt 0x0" ::: "memory");
#endif
        }
        __syncthreads();

        const __bf16* bh_base = (const __bf16*)(smem + (s & 1) * STAGE_BYTES);
        const __bf16* bl_base = bh_base + NB * EDIM;

#pragma unroll
        for (int sub = 0; sub < NB / 16; ++sub) {
            // B 32x16 bf16 layout: lane (n=m, half) holds codebook row n at
            // K k0+8*half .. +7 then k0+16+8*half .. +7 (matches A).
            const __bf16* brh = bh_base + (size_t)(sub * 16 + m) * EDIM + 8 * half;
            const __bf16* brl = bl_base + (size_t)(sub * 16 + m) * EDIM + 8 * half;

            v8f acc0 = {}, acc1 = {}, acc2 = {};   // 3 independent chains (ILP)
#pragma unroll
            for (int kc = 0; kc < 8; ++kc) {
                v16bf bh = vq_load16(brh + kc * 32);
                v16bf bl = vq_load16(brl + kc * 32);
                acc0 = __builtin_amdgcn_wmma_f32_16x16x32_bf16(false, ah[kc], false, bh,
                                                               (short)0, acc0, false, false);
                acc1 = __builtin_amdgcn_wmma_f32_16x16x32_bf16(false, ah[kc], false, bl,
                                                               (short)0, acc1, false, false);
                acc2 = __builtin_amdgcn_wmma_f32_16x16x32_bf16(false, al[kc], false, bh,
                                                               (short)0, acc2, false, false);
            }

            const int code = nb + sub * 16 + m;    // this lane's column
            const float ne = nrm[code];            // 32KB table, L2-hot
            // C/D layout: acc[r] is row (r + 8*half) of this wave's strip.
#pragma unroll
            for (int r = 0; r < 8; ++r) {
                float dot = (acc0[r] + acc1[r]) + acc2[r];
                float sc = fmaf(-2.0f, dot, ne);
                if (sc < minv[r]) { minv[r] = sc; mini[r] = code; }
            }
        }
        __syncthreads();   // all waves done reading before this buffer is reused
    }

    // ---- min+index reduction across the 16 lanes of each half ----
#pragma unroll
    for (int r = 0; r < 8; ++r) {
        float v = minv[r];
        int   i = mini[r];
#pragma unroll
        for (int mask = 1; mask <= 8; mask <<= 1) {
            float ov = __shfl_xor(v, mask, 32);
            int   oi = __shfl_xor(i, mask, 32);
            if (ov < v || (ov == v && oi < i)) { v = ov; i = oi; }
        }
        if (m == 0) idx_out[row_base + 8 * half + r] = i;
    }
}

// --- gather z_q, SSE for loss, histogram, index output ----------------------
__global__ __launch_bounds__(256) void vq_gather(
    const float* __restrict__ z, const float* __restrict__ emb,
    const int* __restrict__ idx, float* __restrict__ out,
    float* __restrict__ cnt, float* __restrict__ sse)
{
    const int n = blockIdx.x;
    const int c = threadIdx.x;
    const int id = idx[n];
    const int b = n >> 10;
    const int p = n & 1023;
    const float e  = emb[(size_t)id * EDIM + c];
    const size_t zo = (size_t)b * 262144 + (size_t)c * 1024 + p;
    const float zv = z[zo];
    out[zo] = e;                       // straight-through: z + sg(z_q - z) == z_q
    float d = e - zv;
    float s = d * d;
#pragma unroll
    for (int mask = 16; mask >= 1; mask >>= 1) s += __shfl_xor(s, mask, 32);
    __shared__ float red[8];
    const int lane = threadIdx.x & 31, w = threadIdx.x >> 5;
    if (lane == 0) red[w] = s;
    __syncthreads();
    if (threadIdx.x == 0) {
        float t = 0.f;
#pragma unroll
        for (int i = 0; i < 8; ++i) t += red[i];
        atomicAdd(sse, t);
        atomicAdd(&cnt[id], 1.0f);
        out[ZQ_ELEMS + 2 + n] = (float)id;   // indices tail (as float)
    }
}

// --- finalize loss + perplexity ---------------------------------------------
__global__ __launch_bounds__(256) void vq_final(const float* __restrict__ cnt,
                                                const float* __restrict__ sse,
                                                float* __restrict__ out) {
    float h = 0.f;
    for (int i = threadIdx.x; i < N_CODE; i += 256) {
        float pm = cnt[i] * (1.0f / 16384.0f);
        h -= pm * logf(pm + 1e-10f);
    }
#pragma unroll
    for (int mask = 16; mask >= 1; mask >>= 1) h += __shfl_xor(h, mask, 32);
    __shared__ float red[8];
    const int lane = threadIdx.x & 31, w = threadIdx.x >> 5;
    if (lane == 0) red[w] = h;
    __syncthreads();
    if (threadIdx.x == 0) {
        float tot = 0.f;
#pragma unroll
        for (int i = 0; i < 8; ++i) tot += red[i];
        out[ZQ_ELEMS]     = 1.25f * sse[0] * (1.0f / (float)ZQ_ELEMS);  // (1+beta)*MSE
        out[ZQ_ELEMS + 1] = expf(tot);                                  // perplexity
    }
}

extern "C" void kernel_launch(void* const* d_in, const int* in_sizes, int n_in,
                              void* d_out, int out_size, void* d_ws, size_t ws_size,
                              hipStream_t stream) {
    const float* z   = (const float*)d_in[0];   // [16,256,32,32]
    const float* emb = (const float*)d_in[1];   // [8192,256]
    float* out = (float*)d_out;                 // z_q | loss | perplexity | indices
    char* ws = (char*)d_ws;

    __bf16* zh  = (__bf16*)(ws + WS_ZH);
    __bf16* zl  = (__bf16*)(ws + WS_ZL);
    __bf16* eh  = (__bf16*)(ws + WS_EH);
    __bf16* el  = (__bf16*)(ws + WS_EL);
    float*  nrm = (float*)(ws + WS_NRM);
    int*    idx = (int*)  (ws + WS_IDX);
    float*  cnt = (float*)(ws + WS_CNT);
    float*  sse = (float*)(ws + WS_SSE);

    vq_convert_z  <<<ZQ_ELEMS / 256, 256, 0, stream>>>(z, zh, zl);
    vq_convert_emb<<<(N_CODE * EDIM) / 256, 256, 0, stream>>>(emb, eh, el);
    vq_norm_emb   <<<N_CODE / 256, 256, 0, stream>>>(emb, nrm);
    vq_init       <<<N_CODE / 256, 256, 0, stream>>>(cnt, sse);
    vq_argmin     <<<N_PIX / 128, 256, 2 * STAGE_BYTES, stream>>>(zh, zl, eh, el, nrm, idx);
    vq_gather     <<<N_PIX, 256, 0, stream>>>(z, emb, idx, out, cnt, sse);
    vq_final      <<<1, 256, 0, stream>>>(cnt, sse, out);
}
